// FHELinear_76613626626116
// MI455X (gfx1250) — 8-bit matrix operand order repaired and checked by reading, compiled, not run on hardware
//
#include <hip/hip_runtime.h>
#include <hip/hip_bf16.h>

typedef int v8i __attribute__((ext_vector_type(8)));
typedef int v4i __attribute__((ext_vector_type(4)));

#define BATCH   65536
#define IN_DIM  256
#define OUT_DIM 256

__global__ __launch_bounds__(256) void quantize_kernel(
    const float* __restrict__ weight, const float* __restrict__ bias,
    const float* __restrict__ in_scale,
    signed char* __restrict__ wh, signed char* __restrict__ wl,
    int* __restrict__ bq, float* __restrict__ out, int out_size)
{
    const int idx = blockIdx.x * 256 + threadIdx.x;
    if (idx < OUT_DIM * IN_DIM) {
        const int q = (int)rintf(weight[idx] * 100.0f);
        wh[idx] = (signed char)(q >> 2);
        wl[idx] = (signed char)(q & 3);
    }
    if (idx < OUT_DIM) {
        bq[idx] = (int)rintf(bias[idx] * (in_scale[0] * 100.0f));
    }
    if (idx == 0) {
        out[(size_t)out_size - 1] = in_scale[0] * 100.0f;
    }
}

__global__ __launch_bounds__(256) void fhe_linear_wmma_kernel(
    const int* __restrict__ x,
    const signed char* __restrict__ wh,
    const signed char* __restrict__ wl,
    const int* __restrict__ bq,
    float* __restrict__ out)
{
    const int lane  = (int)(threadIdx.x & 31u);
    const int wave  = (int)(threadIdx.x >> 5u);
    const int lrow  = lane & 15;
    const int half  = lane >> 4;
    const int nbase = (wave & 3) * 64;
    const int mbase = (int)blockIdx.x * 32 + (wave >> 2) * 16;

    const int* xrow = x + (size_t)(mbase + lrow) * IN_DIM;

    v8i accH[4] = {}, accL[4] = {};

    #pragma unroll
    for (int kt = 0; kt < 4; ++kt) {
        const int k0 = kt * 64;

        int ap[8];
        #pragma unroll
        for (int i = 0; i < 4; ++i) {
            const int kk = k0 + 32 * (i >> 1) + 8 * (i & 1) + 16 * half;
            const v4i x0 = *(const v4i*)(xrow + kk);
            const v4i x1 = *(const v4i*)(xrow + kk + 4);
            ap[2*i+0] = (x0.x) | (x0.y << 8) | (x0.z << 16) | (x0.w << 24);
            ap[2*i+1] = (x1.x) | (x1.y << 8) | (x1.z << 16) | (x1.w << 24);
        }
        const v8i a = { ap[0], ap[1], ap[2], ap[3], ap[4], ap[5], ap[6], ap[7] };

        #pragma unroll
        for (int nt = 0; nt < 4; ++nt) {
            const int ncol = nbase + nt * 16 + lrow;
            const size_t wbase = (size_t)ncol * IN_DIM + (size_t)(k0 + 16 * half);

            const v4i h0 = *(const v4i*)(wh + wbase);
            const v4i h1 = *(const v4i*)(wh + wbase + 32);
            const v4i l0 = *(const v4i*)(wl + wbase);
            const v4i l1 = *(const v4i*)(wl + wbase + 32);
            const v8i bh = __builtin_shufflevector(h0, h1, 0,1,2,3,4,5,6,7);
            const v8i bl = __builtin_shufflevector(l0, l1, 0,1,2,3,4,5,6,7);

            accH[nt] = __builtin_amdgcn_wmma_i32_16x16x64_iu8(
                false, a, true,  bh, accH[nt], false, false);
            accL[nt] = __builtin_amdgcn_wmma_i32_16x16x64_iu8(
                false, a, false, bl, accL[nt], false, false);
        }
    }

    #pragma unroll
    for (int nt = 0; nt < 4; ++nt) {
        const int ncol = nbase + nt * 16 + lrow;
        const int bias = bq[ncol];
        #pragma unroll
        for (int v = 0; v < 8; ++v) {
            const int row = mbase + 8 * half + v;
            const int val = (accH[nt][v] << 2) + accL[nt][v] + bias;
            out[(size_t)row * OUT_DIM + ncol] = (float)val;
        }
    }
}

extern "C" void kernel_launch(void* const* d_in, const int* in_sizes, int n_in,
                              void* d_out, int out_size, void* d_ws, size_t ws_size,
                              hipStream_t stream) {
    const int*   c_x      = (const int*)d_in[0];
    const float* weight   = (const float*)d_in[1];
    const float* bias     = (const float*)d_in[2];
    const float* in_scale = (const float*)d_in[3];
    float*       out      = (float*)d_out;

    signed char* wh = (signed char*)d_ws;
    signed char* wl = wh + (size_t)OUT_DIM * IN_DIM;
    int*         bq = (int*)(wl + (size_t)OUT_DIM * IN_DIM);

    quantize_kernel<<<(OUT_DIM * IN_DIM) / 256, 256, 0, stream>>>(
        weight, bias, in_scale, wh, wl, bq, out, out_size);

    fhe_linear_wmma_kernel<<<BATCH / 32, 256, 0, stream>>>(c_x, wh, wl, bq, out);
}
